// AttnBlock_28097676051223
// MI455X (gfx1250) — matrix operand order repaired and checked by reading, not compiled, not run on hardware
//
#include <hip/hip_runtime.h>

typedef __attribute__((ext_vector_type(16))) _Float16 v16h;
typedef __attribute__((ext_vector_type(8)))  _Float16 v8h;
typedef __attribute__((ext_vector_type(8)))  float    v8f;

constexpr int   BATCH = 4;
constexpr int   CH    = 256;     // channels
constexpr int   HW    = 4096;    // 64*64 spatial
constexpr int   NGRP  = 4;
constexpr int   CG    = CH / NGRP;
constexpr float ATT_SCALE = 0.0625f;   // 1/sqrt(256)
constexpr float GN_EPS    = 1e-6f;

// ---------------------------------------------------------------------------
// WMMA helper (gfx1250 V_WMMA_F32_16X16X32_F16, wave32)
// ---------------------------------------------------------------------------
__device__ __forceinline__ v8f wmma16(v16h a, v16h b, v8f c) {
  return __builtin_amdgcn_wmma_f32_16x16x32_f16(false, a, false, b, (short)0, c,
                                                false, false);
}

// A-matrix 16x32 f16 fragment. Lane holds row = lane%16.
// lanes 0-15: K = {k0..k0+7, k0+16..k0+23}; lanes 16-31: K = {k0+8..k0+15, k0+24..k0+31}
__device__ __forceinline__ v16h load_afrag(const _Float16* __restrict__ base,
                                           int row, int rowstride, int k0, int lane) {
  int kb = (lane & 16) ? 8 : 0;
  const _Float16* p = base + (size_t)row * rowstride + k0 + kb;
  v8h lo = *reinterpret_cast<const v8h*>(p);
  v8h hi = *reinterpret_cast<const v8h*>(p + 16);
  v16h r;
#pragma unroll
  for (int t = 0; t < 8; ++t) { r[t] = lo[t]; r[t + 8] = hi[t]; }
  return r;
}

// B-matrix 32x16 f16 fragment. Lane holds col = lane%16.
// lanes 0-15: K = {k0..k0+7, k0+16..k0+23}; lanes 16-31: K = {k0+8..k0+15, k0+24..k0+31}
__device__ __forceinline__ v16h load_bfrag(const _Float16* __restrict__ base,
                                           int col, int rowstride, int k0, int lane) {
  const _Float16* p = base + (size_t)col * rowstride + k0 + ((lane & 16) ? 8 : 0);
  v8h lo = *reinterpret_cast<const v8h*>(p);
  v8h hi = *reinterpret_cast<const v8h*>(p + 16);
  v16h r;
#pragma unroll
  for (int t = 0; t < 8; ++t) { r[t] = lo[t]; r[t + 8] = hi[t]; }
  return r;
}

// ---------------------------------------------------------------------------
// CDNA5 async global->LDS copy (ASYNCcnt path), per CDNA5 ISA
// ---------------------------------------------------------------------------
__device__ __forceinline__ unsigned lds_off32(const void* p) {
  // LDS aperture keeps the byte offset in addr[31:0]
  return (unsigned)(uintptr_t)p;
}

__device__ __forceinline__ void async_b128(unsigned lds_byte_off, const void* gaddr) {
  asm volatile("global_load_async_to_lds_b128 %0, %1, off"
               :: "v"(lds_byte_off), "v"(gaddr)
               : "memory");
}

__device__ __forceinline__ void wait_async0() {
#if __has_builtin(__builtin_amdgcn_s_wait_asynccnt)
  __builtin_amdgcn_s_wait_asynccnt(0);
#else
  asm volatile("s_wait_asynccnt 0x0" ::: "memory");
#endif
}

// Stage one contiguous 32-row K chunk (32 x 256 f16 = 16KB): 8 b128 per thread
__device__ __forceinline__ void fill_k_async(_Float16* dst_sm,
                                             const _Float16* gsrc, int tid) {
  unsigned dbase = lds_off32(dst_sm);
#pragma unroll
  for (int r = 0; r < 8; ++r) {
    unsigned off = (unsigned)(tid + 128 * r) * 16u;   // bytes
    async_b128(dbase + off, (const char*)gsrc + off);
  }
}

// Stage V chunk: 128 c-rows x 32 j (8KB), global row stride HW: 4 b128 per thread
__device__ __forceinline__ void fill_v_async(_Float16* dst_sm,
                                             const _Float16* vb_c0, int j0, int tid) {
  unsigned dbase = lds_off32(dst_sm);
#pragma unroll
  for (int r = 0; r < 4; ++r) {
    int idx = tid + 128 * r;              // 0..511
    int crow = idx >> 2, part = idx & 3;  // 4 x 16B per c-row
    unsigned loff = (unsigned)(crow * 32 + part * 8) * 2u;
    async_b128(dbase + loff, vb_c0 + (size_t)crow * HW + j0 + part * 8);
  }
}

// ---------------------------------------------------------------------------
// GroupNorm statistics: one block per (batch, group)
// ---------------------------------------------------------------------------
__global__ void gn_stats_kernel(const float* __restrict__ x, float* __restrict__ st) {
  __shared__ float sh[512];
  int bg = blockIdx.x;
  const float* base = x + (size_t)bg * (CG * HW);
  float s = 0.f, ss = 0.f;
  for (int e = threadIdx.x; e < CG * HW; e += 256) {
    float v = base[e]; s += v; ss += v * v;
  }
  sh[threadIdx.x] = s; sh[256 + threadIdx.x] = ss;
  __syncthreads();
  for (int w = 128; w > 0; w >>= 1) {
    if ((int)threadIdx.x < w) {
      sh[threadIdx.x]       += sh[threadIdx.x + w];
      sh[256 + threadIdx.x] += sh[256 + threadIdx.x + w];
    }
    __syncthreads();
  }
  if (threadIdx.x == 0) {
    float n    = (float)(CG * HW);
    float mean = sh[0] / n;
    float var  = sh[256] / n - mean * mean;
    st[bg * 2 + 0] = mean;
    st[bg * 2 + 1] = rsqrtf(var + GN_EPS);
  }
}

// Apply GN, write h channel-last [b][i][c] as f16
__global__ void gn_apply_kernel(const float* __restrict__ x, const float* __restrict__ st,
                                const float* __restrict__ gs, const float* __restrict__ gb,
                                _Float16* __restrict__ h) {
  int idx = blockIdx.x * 256 + threadIdx.x;
  int c = idx & (CH - 1);
  int r = idx >> 8;
  int i = r & (HW - 1);
  int b = r >> 12;
  int g = c >> 6;
  float mean = st[(b * NGRP + g) * 2 + 0];
  float rstd = st[(b * NGRP + g) * 2 + 1];
  float xv = x[((size_t)b * CH + c) * HW + i];
  h[idx] = (_Float16)((xv - mean) * rstd * gs[c] + gb[c]);
}

// fp32 -> f16 weights (wq,wk,wv,wproj concatenated)
__global__ void wcvt_kernel(const float* __restrict__ wq, const float* __restrict__ wk,
                            const float* __restrict__ wv, const float* __restrict__ wp,
                            _Float16* __restrict__ wf) {
  int idx = blockIdx.x * 256 + threadIdx.x;
  int sel = idx >> 16, j = idx & 65535;
  const float* s = (sel == 0) ? wq : (sel == 1) ? wk : (sel == 2) ? wv : wp;
  wf[idx] = (_Float16)s[j];
}

// ---------------------------------------------------------------------------
// QKV GEMM: D[d,i] = sum_c W[d,c] h[c,i] + bias[d]
// wave computes 16(M) x 64(N); q,k stored [i][c] f16; v stored [c][j] f16
// ---------------------------------------------------------------------------
__global__ void qkv_gemm_kernel(const _Float16* __restrict__ wf, const _Float16* __restrict__ h,
                                const float* __restrict__ bq, const float* __restrict__ bk,
                                const float* __restrict__ bv,
                                _Float16* __restrict__ q, _Float16* __restrict__ k,
                                _Float16* __restrict__ v) {
  int lane = threadIdx.x & 31, wave = threadIdx.x >> 5;
  int m0 = (blockIdx.y * 4 + wave) * 16;
  int n0 = blockIdx.x * 64;
  int b = blockIdx.z & 3, mat = blockIdx.z >> 2;
  const _Float16* W  = wf + mat * (CH * CH);
  const _Float16* hb = h + (size_t)b * HW * CH;
  const float* bias = (mat == 0) ? bq : (mat == 1) ? bk : bv;
  v8f acc[4] = {};
  for (int kc = 0; kc < 8; ++kc) {
    int k0 = kc * 32;
    v16h af = load_afrag(W, m0 + (lane & 15), CH, k0, lane);
#pragma unroll
    for (int t = 0; t < 4; ++t) {
      v16h bf = load_bfrag(hb, n0 + t * 16 + (lane & 15), CH, k0, lane);
      acc[t] = wmma16(af, bf, acc[t]);
    }
  }
  int dbase = m0 + 8 * (lane >> 4);
  if (mat < 2) {
    _Float16* out = (mat == 0) ? q : k;
#pragma unroll
    for (int t = 0; t < 4; ++t) {
      int i = n0 + t * 16 + (lane & 15);
      v8h pk;
#pragma unroll
      for (int vv = 0; vv < 8; ++vv) pk[vv] = (_Float16)(acc[t][vv] + bias[dbase + vv]);
      *reinterpret_cast<v8h*>(out + ((size_t)b * HW + i) * CH + dbase) = pk;
    }
  } else {
#pragma unroll
    for (int t = 0; t < 4; ++t) {
      int i = n0 + t * 16 + (lane & 15);
#pragma unroll
      for (int vv = 0; vv < 8; ++vv)
        v[((size_t)b * CH + dbase + vv) * HW + i] = (_Float16)(acc[t][vv] + bias[dbase + vv]);
    }
  }
}

// ---------------------------------------------------------------------------
// Attention pass 1: per 16-row i-tile, sweep all j computing S^T = K_tile x Q,
// online row max & expsum. K chunks (32 rows x 256c = 16KB) staged to LDS by
// async DMA, double-buffered; 4 waves/block share each chunk.
// ---------------------------------------------------------------------------
__global__ void attn_stats_kernel(const _Float16* __restrict__ q,
                                  const _Float16* __restrict__ k,
                                  float* __restrict__ st) {
  __shared__ alignas(128) _Float16 ksm[2][32 * CH];   // 2 x 16KB
  int lane = threadIdx.x & 31, wave = threadIdx.x >> 5;
  int tid  = threadIdx.x;
  int it = blockIdx.x * 4 + wave;
  int b  = blockIdx.y;
  int i0 = it * 16;
  const _Float16* qb = q + (size_t)b * HW * CH;
  const _Float16* kb = k + (size_t)b * HW * CH;
  v16h qf[8];
#pragma unroll
  for (int kc = 0; kc < 8; ++kc)
    qf[kc] = load_bfrag(qb, i0 + (lane & 15), CH, kc * 32, lane);

  fill_k_async(ksm[0], kb, tid);                 // prologue: chunk 0
  float m = -1e30f, lsum = 0.f;
  for (int jc = 0; jc < HW / 32; ++jc) {
    int cur = jc & 1;
    wait_async0();
    __syncthreads();                             // buffer `cur` ready for all waves
    if (jc + 1 < HW / 32)
      fill_k_async(ksm[1 - cur], kb + (size_t)(jc + 1) * 32 * CH, tid);
    const _Float16* ks = ksm[cur];
#pragma unroll
    for (int half = 0; half < 2; ++half) {       // two 16-j sub-tiles
      v8f s = {};
#pragma unroll
      for (int kc = 0; kc < 8; ++kc) {
        v16h kf = load_afrag(ks, half * 16 + (lane & 15), CH, kc * 32, lane);
        s = wmma16(kf, qf[kc], s);
      }
      float mx = -1e30f;
#pragma unroll
      for (int t = 0; t < 8; ++t) mx = fmaxf(mx, s[t] * ATT_SCALE);
      float mnew = fmaxf(m, mx);
      float add = 0.f;
#pragma unroll
      for (int t = 0; t < 8; ++t) add += __expf(s[t] * ATT_SCALE - mnew);
      lsum = lsum * __expf(m - mnew) + add;
      m = mnew;
    }
    __syncthreads();                             // all waves done reading `cur`
  }
  float m2 = __shfl_xor(m, 16, 32);
  float l2 = __shfl_xor(lsum, 16, 32);
  float mt = fmaxf(m, m2);
  float lt = lsum * __expf(m - mt) + l2 * __expf(m2 - mt);
  if (lane < 16) {
    st[((size_t)b * HW + i0 + lane) * 2 + 0] = mt;
    st[((size_t)b * HW + i0 + lane) * 2 + 1] = lt;
  }
}

// ---------------------------------------------------------------------------
// Attention pass 2: recompute S^T per 32-j chunk, P = exp(S*scale - m_row),
// O^T[i,c] += P x V^T. K (16KB) and V (8KB) chunks async-staged to LDS,
// double-buffered, shared by the 4 waves of the block.
// ---------------------------------------------------------------------------
__global__ void attn_av_kernel(const _Float16* __restrict__ q, const _Float16* __restrict__ k,
                               const _Float16* __restrict__ v, const float* __restrict__ st,
                               _Float16* __restrict__ o) {
  __shared__ alignas(128) _Float16 ksm[2][32 * CH];    // 2 x 16KB
  __shared__ alignas(128) _Float16 vsm[2][128 * 32];   // 2 x 8KB, [crow][j]
  int lane = threadIdx.x & 31, wave = threadIdx.x >> 5;
  int tid  = threadIdx.x;
  int it = blockIdx.x * 4 + wave;
  int b  = blockIdx.y;
  int c0 = blockIdx.z * 128;
  int i0 = it * 16;
  const _Float16* qb = q + (size_t)b * HW * CH;
  const _Float16* kb = k + (size_t)b * HW * CH;
  const _Float16* vb = v + (size_t)b * CH * HW + (size_t)c0 * HW;  // [c][j], c rel
  v16h qf[8];
#pragma unroll
  for (int kc = 0; kc < 8; ++kc)
    qf[kc] = load_bfrag(qb, i0 + (lane & 15), CH, kc * 32, lane);
  float mrow = st[((size_t)b * HW + i0 + (lane & 15)) * 2 + 0];
  float lrow = st[((size_t)b * HW + i0 + (lane & 15)) * 2 + 1];
  v8f oacc[8] = {};

  fill_k_async(ksm[0], kb, tid);
  fill_v_async(vsm[0], vb, 0, tid);
  for (int jc = 0; jc < HW / 32; ++jc) {
    int cur = jc & 1;
    wait_async0();
    __syncthreads();
    if (jc + 1 < HW / 32) {
      fill_k_async(ksm[1 - cur], kb + (size_t)(jc + 1) * 32 * CH, tid);
      fill_v_async(vsm[1 - cur], vb, (jc + 1) * 32, tid);
    }
    const _Float16* ks = ksm[cur];
    const _Float16* vs = vsm[cur];
    v8f s0 = {}, s1 = {};
#pragma unroll
    for (int kc = 0; kc < 8; ++kc) {
      v16h kf = load_afrag(ks, (lane & 15), CH, kc * 32, lane);
      s0 = wmma16(kf, qf[kc], s0);
    }
#pragma unroll
    for (int kc = 0; kc < 8; ++kc) {
      v16h kf = load_afrag(ks, 16 + (lane & 15), CH, kc * 32, lane);
      s1 = wmma16(kf, qf[kc], s1);
    }
    v16h pa;
#pragma unroll
    for (int t = 0; t < 8; ++t) pa[t]     = (_Float16)__expf(s0[t] * ATT_SCALE - mrow);
#pragma unroll
    for (int t = 0; t < 8; ++t) pa[8 + t] = (_Float16)__expf(s1[t] * ATT_SCALE - mrow);
#pragma unroll
    for (int n = 0; n < 8; ++n) {
      v16h vf = load_bfrag(vs, n * 16 + (lane & 15), 32, 0, lane);
      oacc[n] = wmma16(pa, vf, oacc[n]);
    }
    __syncthreads();
  }
  float linv[8];
#pragma unroll
  for (int vv = 0; vv < 8; ++vv)
    linv[vv] = 1.0f / __shfl(lrow, vv + 8 * (lane >> 4), 32);
#pragma unroll
  for (int n = 0; n < 8; ++n) {
    int c = c0 + n * 16 + (lane & 15);
#pragma unroll
    for (int vv = 0; vv < 8; ++vv) {
      int i = i0 + vv + 8 * (lane >> 4);
      o[((size_t)b * HW + i) * CH + c] = (_Float16)(oacc[n][vv] * linv[vv]);
    }
  }
}

// ---------------------------------------------------------------------------
// Proj GEMM + bias + residual: out[b,d,i] = sum_c Wp[d,c] o[c,i] + bp[d] + x
// ---------------------------------------------------------------------------
__global__ void proj_gemm_kernel(const _Float16* __restrict__ wf, const _Float16* __restrict__ o,
                                 const float* __restrict__ bp, const float* __restrict__ x,
                                 float* __restrict__ out) {
  int lane = threadIdx.x & 31, wave = threadIdx.x >> 5;
  int m0 = (blockIdx.y * 4 + wave) * 16;
  int n0 = blockIdx.x * 64;
  int b  = blockIdx.z;
  const _Float16* W  = wf + 3 * CH * CH;
  const _Float16* ob = o + (size_t)b * HW * CH;
  v8f acc[4] = {};
  for (int kc = 0; kc < 8; ++kc) {
    int k0 = kc * 32;
    v16h af = load_afrag(W, m0 + (lane & 15), CH, k0, lane);
#pragma unroll
    for (int t = 0; t < 4; ++t) {
      v16h bf = load_bfrag(ob, n0 + t * 16 + (lane & 15), CH, k0, lane);
      acc[t] = wmma16(af, bf, acc[t]);
    }
  }
  int dbase = m0 + 8 * (lane >> 4);
#pragma unroll
  for (int t = 0; t < 4; ++t) {
    int i = n0 + t * 16 + (lane & 15);
#pragma unroll
    for (int vv = 0; vv < 8; ++vv) {
      int d = dbase + vv;
      size_t idx = ((size_t)b * CH + d) * HW + i;
      out[idx] = acc[t][vv] + bp[d] + x[idx];
    }
  }
}

// ---------------------------------------------------------------------------
extern "C" void kernel_launch(void* const* d_in, const int* in_sizes, int n_in,
                              void* d_out, int out_size, void* d_ws, size_t ws_size,
                              hipStream_t stream) {
  const float* x  = (const float*)d_in[0];
  const float* gs = (const float*)d_in[1];
  const float* gb = (const float*)d_in[2];
  const float* wq = (const float*)d_in[3];
  const float* bq = (const float*)d_in[4];
  const float* wk = (const float*)d_in[5];
  const float* bk = (const float*)d_in[6];
  const float* wv = (const float*)d_in[7];
  const float* bv = (const float*)d_in[8];
  const float* wp = (const float*)d_in[9];
  const float* bp = (const float*)d_in[10];
  float* out = (float*)d_out;
  (void)in_sizes; (void)n_in; (void)out_size; (void)ws_size;

  char* base = (char*)d_ws;
  size_t off = 0;
  auto take = [&](size_t bytes) -> void* {
    void* p = base + off;
    off += (bytes + 511) & ~(size_t)511;
    return p;
  };
  float*    gnst = (float*)take((size_t)BATCH * NGRP * 2 * sizeof(float));
  float*    atst = (float*)take((size_t)BATCH * HW * 2 * sizeof(float));
  _Float16* wf   = (_Float16*)take((size_t)4 * CH * CH * sizeof(_Float16));
  _Float16* h    = (_Float16*)take((size_t)BATCH * HW * CH * sizeof(_Float16));
  _Float16* q    = (_Float16*)take((size_t)BATCH * HW * CH * sizeof(_Float16));
  _Float16* k    = (_Float16*)take((size_t)BATCH * HW * CH * sizeof(_Float16));
  _Float16* v    = (_Float16*)take((size_t)BATCH * HW * CH * sizeof(_Float16));
  _Float16* o    = (_Float16*)take((size_t)BATCH * HW * CH * sizeof(_Float16));

  gn_stats_kernel<<<BATCH * NGRP, 256, 0, stream>>>(x, gnst);
  wcvt_kernel<<<(4 * CH * CH) / 256, 256, 0, stream>>>(wq, wk, wv, wp, wf);
  gn_apply_kernel<<<(BATCH * HW * CH) / 256, 256, 0, stream>>>(x, gnst, gs, gb, h);
  qkv_gemm_kernel<<<dim3(HW / 64, 4, 12), 128, 0, stream>>>(wf, h, bq, bk, bv, q, k, v);
  attn_stats_kernel<<<dim3(HW / 64, BATCH), 128, 0, stream>>>(q, k, atst);
  attn_av_kernel<<<dim3(HW / 64, BATCH, 2), 128, 0, stream>>>(q, k, v, atst, o);
  proj_gemm_kernel<<<dim3(HW / 64, 4, BATCH), 128, 0, stream>>>(wf, o, bp, x, out);
}
